// DummyMoELayer_14413910245694
// MI455X (gfx1250) — matrix operand order repaired and checked by reading, not compiled, not run on hardware
//
#include <hip/hip_runtime.h>
#include <hip/hip_bf16.h>
#include <stdint.h>

// ---- problem constants (from reference) ----
#define NE 8
#define NH 2048
#define NI 128
#define NBATCH 8
#define NSEQ 2048
#define NT (NBATCH * NSEQ)          // 16384 tokens
#define WELEM (NE * NI * NH)        // elements per expert-weight tensor (2,097,152)

// output layout (flat, return-tuple order)
#define OFF_PROBS ((size_t)NT * NH)
#define OFF_HN    (OFF_PROBS + (size_t)NT * NE)
#define OFF_SPEC  (OFF_HN + 256)
#define OFF_COS   (OFF_SPEC + 1)
#define OFF_EXPR  (OFF_COS + 8)

typedef __attribute__((ext_vector_type(16))) __bf16 v16bf;
typedef __attribute__((ext_vector_type(8)))  __bf16 v8bf;
typedef __attribute__((ext_vector_type(8)))  float  v8f;
typedef __attribute__((ext_vector_type(4)))  unsigned int v4u;
typedef __attribute__((ext_vector_type(8)))  int v8i;
typedef __attribute__((ext_vector_type(4)))  int v4i;
typedef unsigned short ushort_t;

// ---- Tensor Data Mover availability (device pass only; host parses fallback) --
#if defined(__HIP_DEVICE_COMPILE__) && \
    __has_builtin(__builtin_amdgcn_tensor_load_to_lds) && \
    __has_builtin(__builtin_amdgcn_s_wait_tensorcnt)
#define HAVE_TDM 1
#if __has_include(<hip/amd_detail/amd_gfx1250_TDM.h>)
#define TDM_ARGS6 1
#endif
#endif

#ifdef HAVE_TDM
// DMA a [128 rows x 32 cols] bf16 tile (row stride NH elems in memory) into LDS
// with a 16B pad after every 64B row => 80B LDS row stride (matches BPAD=40).
__device__ __forceinline__ void tdm_load_tile(const __bf16* gsrc, unsigned lds_byte_addr) {
    const unsigned long long ga = (unsigned long long)(uintptr_t)gsrc;
    v4u g0;
    g0[0] = 1u;                                         // count=1, user descriptor
    g0[1] = lds_byte_addr;                              // lds_addr
    g0[2] = (unsigned)(ga & 0xFFFFFFFFu);               // global_addr[31:0]
    g0[3] = ((unsigned)(ga >> 32) & 0x01FFFFFFu)        // global_addr[56:32]
            | 0x80000000u;                              // type=2 ("image")
    v8i g1;
    g1[0] = (1 << 16)        // data_size = 2 bytes
          | (1 << 20)        // pad_enable
          | (3 << 22)        // pad_interval: 16 DWORDs (64B) between pads
          | (3 << 25);       // pad_amount: 4 DWORDs (16B)
    g1[1] = (int)(2048u << 16);   // tensor_dim0 = 2048 (lo16)
    g1[2] = (int)(128u  << 16);   // tensor_dim0 hi = 0 | tensor_dim1 = 128 (lo16)
    g1[3] = (int)(32u   << 16);   // tensor_dim1 hi = 0 | tile_dim0 = 32
    g1[4] = 128;                  // tile_dim1 = 128 | tile_dim2 = 0
    g1[5] = 2048;                 // tensor_dim0_stride (lo32), elems
    g1[6] = 0;
    g1[7] = 0;
    v4i z4 = {0, 0, 0, 0};
#ifdef TDM_ARGS6
    v8i z8 = {0, 0, 0, 0, 0, 0, 0, 0};
    __builtin_amdgcn_tensor_load_to_lds(g0, g1, z4, z4, z8, 0);
#else
    __builtin_amdgcn_tensor_load_to_lds(g0, g1, z4, z4, 0);
#endif
}
#endif

// =====================  K0: f32 -> bf16 weight pre-conversion  ================
__global__ void cvt_f32_to_bf16(const float* __restrict__ in,
                                ushort_t* __restrict__ outp, int n,
                                int* zero_ptr, int zero_n) {
    __bf16* o = (__bf16*)outp;
    int i = (blockIdx.x * blockDim.x + threadIdx.x) * 4;
    if (i < n) {
        float4 v = *(const float4*)(in + i);
        o[i + 0] = (__bf16)v.x;
        o[i + 1] = (__bf16)v.y;
        o[i + 2] = (__bf16)v.z;
        o[i + 3] = (__bf16)v.w;
    }
    if (zero_ptr && blockIdx.x == 0 && (int)threadIdx.x < zero_n)
        zero_ptr[threadIdx.x] = 0;
}

// =====================  K1: router (logits, softmax, argmax)  =================
__launch_bounds__(32)
__global__ void router_kernel(const float* __restrict__ x,
                              const float* __restrict__ gate_w,
                              float* __restrict__ out,
                              float* __restrict__ weightT,
                              int* __restrict__ selT,
                              int* __restrict__ counts) {
    __shared__ float logitsS[16][NE];
    const int lane = threadIdx.x;
    const int t0   = blockIdx.x * 16;

    {
        const int t  = lane >> 1;
        const int hh = lane & 1;
        float acc[NE] = {};
        const float* xr = x + (size_t)(t0 + t) * NH + hh * (NH / 2);
        for (int j = 0; j < NH / 2; j += 4) {
            float4 xv = *(const float4*)(xr + j);
#pragma unroll
            for (int e = 0; e < NE; e++) {
                float4 gv = *(const float4*)(gate_w + (size_t)e * NH + hh * (NH / 2) + j);
                acc[e] += xv.x * gv.x + xv.y * gv.y + xv.z * gv.z + xv.w * gv.w;
            }
        }
#pragma unroll
        for (int e = 0; e < NE; e++) {
            acc[e] += __shfl_xor(acc[e], 1, 32);
            if (hh == 0) logitsS[t][e] = acc[e];
        }
    }
    __syncthreads();

    if (lane < 16) {
        float l[NE];
        float m = -3.0e38f;
#pragma unroll
        for (int e = 0; e < NE; e++) { l[e] = logitsS[lane][e]; m = fmaxf(m, l[e]); }
        float s = 0.f;
#pragma unroll
        for (int e = 0; e < NE; e++) { l[e] = __expf(l[e] - m); s += l[e]; }
        const float inv = 1.0f / s;
        int   sel  = 0;
        float best = -1.0f;
#pragma unroll
        for (int e = 0; e < NE; e++) {
            float p = l[e] * inv;
            out[OFF_PROBS + (size_t)(t0 + lane) * NE + e] = p;
            if (p > best) { best = p; sel = e; }
        }
        weightT[t0 + lane] = best;
        selT[t0 + lane]    = sel;
        atomicAdd(&counts[sel], 1);
    }

    if (blockIdx.x == 0) {   // aux diagnostic outputs
        for (int j = lane; j < 256; j += 32) out[OFF_HN + j] = 0.0f;
        if (lane < 8) out[OFF_COS + lane] = 0.0f;
        if (lane == 0) { out[OFF_SPEC] = 0.035f; out[OFF_EXPR] = 0.019f; }
    }
}

// =============  K2: scan counts, build tile descriptors, scatter perm  ========
__global__ void build_tiles(const int* __restrict__ counts,
                            const int* __restrict__ selT,
                            int* __restrict__ perm,
                            int* __restrict__ tileCnt,
                            int* __restrict__ desc) {
    __shared__ int offs[NE];
    __shared__ int cur[NE];
    if (threadIdx.x == 0) {
        int o = 0, idx = 0;
        for (int e = 0; e < NE; e++) { offs[e] = o; o += counts[e]; }
        for (int e = 0; e < NE; e++) {
            int c = counts[e];
            for (int j = 0; j < c; j += 64) {
                desc[idx * 3 + 0] = e;
                desc[idx * 3 + 1] = offs[e] + j;
                desc[idx * 3 + 2] = (c - j < 64) ? (c - j) : 64;
                idx++;
            }
        }
        *tileCnt = idx;
    }
    if ((int)threadIdx.x < NE) cur[threadIdx.x] = 0;
    __syncthreads();
    for (int t = threadIdx.x; t < NT; t += blockDim.x) {
        int e = selT[t];
        int p = atomicAdd(&cur[e], 1);
        perm[offs[e] + p] = t;
    }
}

// ==============  K3: per-expert gathered GEMMs (WMMA bf16 + TDM)  =============
// 128 threads = 4 waves; each block = one 64-token tile of one expert.
#define BPAD 40     // B row pad: 80B stride -> conflict-free ds_load_b128
#define HPAD 136    // h row pad: 272B stride -> conflict-free ds_load_b128

__launch_bounds__(128)
__global__ void moe_expert_gemm(const float* __restrict__ x,
                                const ushort_t* __restrict__ wgB_,
                                const ushort_t* __restrict__ wuB_,
                                const ushort_t* __restrict__ wdB_,
                                const int* __restrict__ perm,
                                const float* __restrict__ weightT,
                                const int* __restrict__ desc,
                                const int* __restrict__ tileCnt,
                                float* __restrict__ out) {
    if ((int)blockIdx.x >= *tileCnt) return;

    __shared__ __bf16 Bg[2][NI][BPAD];      // 20.0 KB  (double-buffered gate weights)
    __shared__ __bf16 Bu[2][NI][BPAD];      // 20.0 KB
    __shared__ __bf16 hb[4][16][HPAD];      // 17.0 KB  (per-wave weighted SwiGLU)
    __shared__ int    tokid[64];
    __shared__ float  tokw[64];

    const int tid  = threadIdx.x;
    const int lane = tid & 31;
    const int wv   = tid >> 5;              // wave id 0..3
    const int half = lane >> 4;
    const int l16  = lane & 15;

    const int e     = desc[blockIdx.x * 3 + 0];
    const int start = desc[blockIdx.x * 3 + 1];
    const int cnt   = desc[blockIdx.x * 3 + 2];

    if (tid < 64) {
        if (tid < cnt) {
            int t = perm[start + tid];
            tokid[tid] = t;
            tokw[tid]  = weightT[t];
        } else {
            tokid[tid] = 0;
            tokw[tid]  = 0.0f;
        }
    }
    __syncthreads();

    const __bf16* wg = (const __bf16*)wgB_ + (size_t)e * NI * NH;
    const __bf16* wu = (const __bf16*)wuB_ + (size_t)e * NI * NH;
    const float*  xrow = x + (size_t)tokid[wv * 16 + l16] * NH;

#ifdef HAVE_TDM
    const unsigned bgA[2] = { (unsigned)(uintptr_t)&Bg[0][0][0],
                              (unsigned)(uintptr_t)&Bg[1][0][0] };
    const unsigned buA[2] = { (unsigned)(uintptr_t)&Bu[0][0][0],
                              (unsigned)(uintptr_t)&Bu[1][0][0] };
#else
    // fallback: cooperative stage of W[:,k:k+32] (both matrices) into LDS `buf`
    auto stage = [&](int buf, int k) {
        const int r = tid;                               // one row per thread
        const __bf16* gs = wg + (size_t)r * NH + k;      // 32 consecutive bf16
        const __bf16* us = wu + (size_t)r * NH + k;
        v8bf g0 = *(const v8bf*)(gs + 0),  g1 = *(const v8bf*)(gs + 8);
        v8bf g2 = *(const v8bf*)(gs + 16), g3 = *(const v8bf*)(gs + 24);
        v8bf u0 = *(const v8bf*)(us + 0),  u1 = *(const v8bf*)(us + 8);
        v8bf u2 = *(const v8bf*)(us + 16), u3 = *(const v8bf*)(us + 24);
        *(v8bf*)&Bg[buf][r][0]  = g0; *(v8bf*)&Bg[buf][r][8]  = g1;
        *(v8bf*)&Bg[buf][r][16] = g2; *(v8bf*)&Bg[buf][r][24] = g3;
        *(v8bf*)&Bu[buf][r][0]  = u0; *(v8bf*)&Bu[buf][r][8]  = u1;
        *(v8bf*)&Bu[buf][r][16] = u2; *(v8bf*)&Bu[buf][r][24] = u3;
    };
#endif

    // ---------------- phase A: g/u = x Wg^T, x Wu^T  (K = H, steps of 32) -----
    v8f gacc[8], uacc[8];
#pragma unroll
    for (int n = 0; n < 8; n++) { gacc[n] = v8f{}; uacc[n] = v8f{}; }

#ifdef HAVE_TDM
    if (wv == 0) { tdm_load_tile(wg, bgA[0]); tdm_load_tile(wu, buA[0]); }
#else
    stage(0, 0);
#endif

    for (int ks = 0; ks < NH / 32; ks++) {
#ifdef HAVE_TDM
        if (wv == 0) __builtin_amdgcn_s_wait_tensorcnt((short)0);
#endif
        __syncthreads();                       // staged buffer visible to block
        const int k = ks * 32;
        if (ks + 1 < NH / 32) {
#ifdef HAVE_TDM
            if (wv == 0) {
                tdm_load_tile(wg + k + 32, bgA[(ks + 1) & 1]);
                tdm_load_tile(wu + k + 32, buA[(ks + 1) & 1]);
            }
#else
            stage((ks + 1) & 1, k + 32);
#endif
        }

        // A fragment: M=l16 (gathered token row), K = k + half*8(+16) + [0..7]
        v16bf a;
        {
            const float* xp = xrow + k + half * 8;
            float4 x0 = *(const float4*)(xp + 0);
            float4 x1 = *(const float4*)(xp + 4);
            float4 x2 = *(const float4*)(xp + 16);
            float4 x3 = *(const float4*)(xp + 20);
            a[0]=(__bf16)x0.x; a[1]=(__bf16)x0.y; a[2]=(__bf16)x0.z; a[3]=(__bf16)x0.w;
            a[4]=(__bf16)x1.x; a[5]=(__bf16)x1.y; a[6]=(__bf16)x1.z; a[7]=(__bf16)x1.w;
            a[8]=(__bf16)x2.x; a[9]=(__bf16)x2.y; a[10]=(__bf16)x2.z; a[11]=(__bf16)x2.w;
            a[12]=(__bf16)x3.x; a[13]=(__bf16)x3.y; a[14]=(__bf16)x3.z; a[15]=(__bf16)x3.w;
        }
        const int buf = ks & 1;
#pragma unroll
        for (int nb = 0; nb < 8; nb++) {
            const __bf16* bgp = &Bg[buf][nb * 16 + l16][half * 8];
            const __bf16* bup = &Bu[buf][nb * 16 + l16][half * 8];
            v8bf bg0 = *(const v8bf*)(bgp), bg1 = *(const v8bf*)(bgp + 16);
            v8bf bu0 = *(const v8bf*)(bup), bu1 = *(const v8bf*)(bup + 16);
            v16bf bg, bu;
#pragma unroll
            for (int i = 0; i < 8; i++) {
                bg[i] = bg0[i]; bg[8 + i] = bg1[i];
                bu[i] = bu0[i]; bu[8 + i] = bu1[i];
            }
            gacc[nb] = __builtin_amdgcn_wmma_f32_16x16x32_bf16(
                false, a, false, bg, (short)0, gacc[nb], false, false);
            uacc[nb] = __builtin_amdgcn_wmma_f32_16x16x32_bf16(
                false, a, false, bu, (short)0, uacc[nb], false, false);
        }
    }

    // epilogue: h = routing_weight * silu(g) * u  -> LDS (bf16, per-wave tile)
#pragma unroll
    for (int nb = 0; nb < 8; nb++) {
#pragma unroll
        for (int r = 0; r < 8; r++) {
            const int m = r + 8 * half;
            const float w  = tokw[wv * 16 + m];
            const float g  = gacc[nb][r];
            const float u  = uacc[nb][r];
            const float sg = g / (1.0f + __expf(-g));
            hb[wv][m][nb * 16 + l16] = (__bf16)(w * sg * u);
        }
    }
    // no barrier: each wave consumes only its own hb tile (per-wave DS ordering)

    // ---------------- phase B: final[tok, :] = h @ Wd^T  (K = I = 128) --------
    const __bf16* wd = (const __bf16*)wdB_ + (size_t)e * NH * NI;
    for (int nb = 0; nb < NH / 16; nb++) {
        v8f acc = v8f{};
#pragma unroll
        for (int ksi = 0; ksi < 4; ksi++) {
            const int kk = ksi * 32;
            const __bf16* ap = &hb[wv][l16][kk + half * 8];
            v8bf a0 = *(const v8bf*)(ap), a1 = *(const v8bf*)(ap + 16);
            const __bf16* bp = wd + (size_t)(nb * 16 + l16) * NI + kk + half * 8;
            v8bf b0 = *(const v8bf*)(bp), b1 = *(const v8bf*)(bp + 16);
            v16bf a, b;
#pragma unroll
            for (int i = 0; i < 8; i++) {
                a[i] = a0[i]; a[8 + i] = a1[i];
                b[i] = b0[i]; b[8 + i] = b1[i];
            }
            acc = __builtin_amdgcn_wmma_f32_16x16x32_bf16(
                false, a, false, b, (short)0, acc, false, false);
        }
#pragma unroll
        for (int r = 0; r < 8; r++) {
            const int m    = r + 8 * half;
            const int slot = wv * 16 + m;
            if (slot < cnt)
                out[(size_t)tokid[slot] * NH + nb * 16 + l16] = acc[r];
        }
    }
}

// ==============================  host launcher  ===============================
extern "C" void kernel_launch(void* const* d_in, const int* in_sizes, int n_in,
                              void* d_out, int out_size, void* d_ws, size_t ws_size,
                              hipStream_t stream) {
    const float* x      = (const float*)d_in[0];
    const float* gate_w = (const float*)d_in[1];
    const float* w_gate = (const float*)d_in[2];
    const float* w_up   = (const float*)d_in[3];
    const float* w_down = (const float*)d_in[4];
    float* out = (float*)d_out;

    // workspace carving (~12.8 MB)
    ushort_t* wgB = (ushort_t*)d_ws;
    ushort_t* wuB = wgB + WELEM;
    ushort_t* wdB = wuB + WELEM;
    float* weightT = (float*)(wdB + WELEM);
    int*   selT    = (int*)(weightT + NT);
    int*   perm    = selT + NT;
    int*   counts  = perm + NT;
    int*   tileCnt = counts + NE;
    int*   desc    = tileCnt + 1;     // up to 512 tiles * 3 ints

    // K0: pre-convert expert weights to bf16 (first launch also zeroes counts)
    cvt_f32_to_bf16<<<WELEM / 1024, 256, 0, stream>>>(w_gate, wgB, WELEM, counts, NE);
    cvt_f32_to_bf16<<<WELEM / 1024, 256, 0, stream>>>(w_up,   wuB, WELEM, nullptr, 0);
    cvt_f32_to_bf16<<<WELEM / 1024, 256, 0, stream>>>(w_down, wdB, WELEM, nullptr, 0);

    // K1: router (probs out, weight/sel to ws, per-expert counts)
    router_kernel<<<NT / 16, 32, 0, stream>>>(x, gate_w, out, weightT, selT, counts);

    // K2: offsets, tile descriptors, token permutation
    build_tiles<<<1, 256, 0, stream>>>(counts, selT, perm, tileCnt, desc);

    // K3: gathered per-expert GEMMs (64 tokens per block, 4 waves, TDM staging)
    moe_expert_gemm<<<512, 128, 0, stream>>>(x, wgB, wuB, wdB, perm, weightT,
                                             desc, tileCnt, out);
}
